// LstmNetFullOpti_57372173140196
// MI455X (gfx1250) — matrix operand order repaired and checked by reading, not compiled, not run on hardware
//
#include <hip/hip_runtime.h>
#include <hip/hip_bf16.h>
#include <stdint.h>

#define N_   1024
#define A_   128
#define T_   50
#define TA_  6400      // T_*A_
#define G4_  512       // 4*A_
#define KC_  256       // concat K = [xt | h]

typedef __attribute__((ext_vector_type(16))) __bf16   v16bf;
typedef __attribute__((ext_vector_type(8)))  float    v8f;
typedef __attribute__((ext_vector_type(4)))  uint32_t u32x4;
typedef __attribute__((ext_vector_type(8)))  uint32_t u32x8;

__device__ __forceinline__ __bf16 f2bf(float x) { return (__bf16)x; }
__device__ __forceinline__ v8f zero8() { v8f z = {0.f,0.f,0.f,0.f,0.f,0.f,0.f,0.f}; return z; }

__device__ __forceinline__ v8f wmma_bf16(v16bf a, v16bf b, v8f c) {
  // (neg_a, A, neg_b, B, c_mod, C, reuse_a, reuse_b)
  return __builtin_amdgcn_wmma_f32_16x16x32_bf16(false, a, false, b, (short)0, c, false, false);
}

// ---- Tensor Data Mover: async 2D tile (tile_dim0=128 bf16 x tile_dim1=50 rows)
// from global (row stride 128 elems) into LDS at lds_off.  D# per ISA 8.3/8.4,
// 2-group form (VADDR2/VADDR3 = NULL => up-to-2D tensor). Tracked by TENSORcnt.
__device__ __forceinline__ void tdm_load_xt(const __bf16* gsrc, uint32_t lds_off) {
  uint64_t ga = (uint64_t)(size_t)gsrc;
  u32x4 g0;
  g0[0] = 1u;                                   // count=1, user descriptor
  g0[1] = lds_off;                              // lds_addr (bytes)
  g0[2] = (uint32_t)ga;                         // global_addr[31:0] (tile start)
  g0[3] = (uint32_t)(ga >> 32) | (2u << 30);    // global_addr[56:32] | type=2
  u32x8 g1;
  g1[0] = 0x00010000u;                          // wg_mask=0 | data_size=1 (2 bytes)
  g1[1] = (uint32_t)(128u << 16);               // tensor_dim0 = 128 (lo16)
  g1[2] = (uint32_t)(50u << 16);                // tensor_dim0 hi | tensor_dim1 = 50
  g1[3] = (uint32_t)(128u << 16);               // tensor_dim1 hi | tile_dim0 = 128
  g1[4] = 50u;                                  // tile_dim1 = 50 | tile_dim2 = 0
  g1[5] = 128u;                                 // tensor_dim0_stride = 128 (lo32)
  g1[6] = 0u;                                   // stride hi | tensor_dim1_stride lo
  g1[7] = 0u;
  asm volatile("tensor_load_to_lds %0, %1" :: "s"(g0), "s"(g1) : "memory");
}

// ---- WMMA fragment loaders (wave32; contiguous k = 16*(lane/16) + i) -------------------
__device__ __forceinline__ v16bf load_a_frag(const __bf16* buf, int ld, int row0, int k0) {
  const int lane = threadIdx.x & 31;
  const int r    = row0 + (lane & 15);
  const int koff = (lane >> 4) ? 16 : 0;
  const __bf16* p = buf + r * ld + k0 + koff;
  v16bf a;
#pragma unroll
  for (int i = 0; i < 16; ++i) a[i] = p[i];
  return a;
}
__device__ __forceinline__ v16bf load_a_frag_tr(const __bf16* buf, int ld, int row0, int k0) {
  const int lane = threadIdx.x & 31;
  const int r    = row0 + (lane & 15);
  const int koff = (lane >> 4) ? 16 : 0;
  v16bf a;
#pragma unroll
  for (int i = 0; i < 16; ++i) {
    a[i]     = buf[(k0 + koff + i) * ld + r];
  }
  return a;
}
__device__ __forceinline__ v16bf load_b_frag_wt(const __bf16* w, int ldw, int col0, int k0) {
  const int lane = threadIdx.x & 31;
  const int c    = col0 + (lane & 15);
  const int koff = (lane >> 4) ? 16 : 0;
  const __bf16* p = w + c * ldw + k0 + koff;
  v16bf b;
#pragma unroll
  for (int i = 0; i < 16; ++i) b[i] = p[i];
  return b;
}
__device__ __forceinline__ v16bf load_b_frag_rm(const __bf16* buf, int ld, int col0, int k0) {
  const int lane = threadIdx.x & 31;
  const int c    = col0 + (lane & 15);
  const int koff = (lane >> 4) ? 16 : 0;
  v16bf b;
#pragma unroll
  for (int i = 0; i < 16; ++i) b[i] = buf[(k0 + koff + i) * ld + c];
  return b;
}

// ---- wave32 reductions -----------------------------------------------------
__device__ __forceinline__ float wsum(float x) {
#pragma unroll
  for (int m = 16; m >= 1; m >>= 1) x += __shfl_xor(x, m, 32);
  return x;
}
__device__ __forceinline__ float wmaxr(float x) {
#pragma unroll
  for (int m = 16; m >= 1; m >>= 1) x = fmaxf(x, __shfl_xor(x, m, 32));
  return x;
}
__device__ __forceinline__ float wminr(float x) {
#pragma unroll
  for (int m = 16; m >= 1; m >>= 1) x = fminf(x, __shfl_xor(x, m, 32));
  return x;
}
__device__ __forceinline__ float sigm(float x) { return 1.0f / (1.0f + __expf(-x)); }
__device__ __forceinline__ float clip1(float x) { return fminf(fmaxf(x, -1.f), 1.f); }

// ---- prep kernels ----------------------------------------------------------
__global__ void k_prep_wcat(const float* __restrict__ wih, const float* __restrict__ whh,
                            __bf16* __restrict__ wcat) {
  int idx = blockIdx.x * 256 + threadIdx.x;
  if (idx >= G4_ * KC_) return;
  int j = idx >> 8, k = idx & 255;
  float v = (k < A_) ? wih[j * A_ + k] : whh[j * A_ + (k - A_)];
  wcat[idx] = f2bf(v);
}
__global__ void k_prep_wlin(const float* __restrict__ wl, __bf16* __restrict__ wlb) {
  int idx = blockIdx.x * 256 + threadIdx.x;
  if (idx < A_ * TA_) wlb[idx] = f2bf(wl[idx]);
}
__global__ void k_prep_csum(const float* __restrict__ wl, float* __restrict__ cs) {
  int idx = blockIdx.x * 256 + threadIdx.x;
  if (idx >= A_ * A_) return;
  int j = idx >> 7, a = idx & 127;
  float s = 0.f;
  for (int t = 0; t < T_; ++t) s += wl[(size_t)j * TA_ + t * A_ + a];
  cs[idx] = s;
}

// ---- LayerNorm over (T,A) per sample; emits bf16 seq for the TDM path -----
__global__ __launch_bounds__(256) void k_layernorm(const float* __restrict__ x,
                                                   const float* __restrict__ gamma,
                                                   const float* __restrict__ beta,
                                                   __bf16* __restrict__ seqbf) {
  __shared__ float r1[256], r2[256];
  const int n = blockIdx.x, tid = threadIdx.x;
  const float* xp = x + (size_t)n * TA_;
  float s = 0.f, s2 = 0.f;
  for (int i = tid; i < TA_; i += 256) { float v = xp[i]; s += v; s2 += v * v; }
  r1[tid] = s; r2[tid] = s2; __syncthreads();
  for (int off = 128; off; off >>= 1) {
    if (tid < off) { r1[tid] += r1[tid + off]; r2[tid] += r2[tid + off]; }
    __syncthreads();
  }
  float mu   = r1[0] * (1.0f / TA_);
  float var  = r2[0] * (1.0f / TA_) - mu * mu;
  float rstd = rsqrtf(var + 1e-5f);
  float g = gamma[0], b = beta[0];
  __bf16* sp = seqbf + (size_t)n * TA_;
  for (int i = tid; i < TA_; i += 256) sp[i] = f2bf((xp[i] - mu) * rstd * g + b);
}

// ---- persistent single-WGP LSTM: 1024 serial steps --------------------------
// LDS: Wcat bf16 512x256 (256K) | Xh 64x128 (16K) | Xx ping/pong 2x 64x128 (32K)
// c-state and biases live in per-lane registers (fixed cell ownership).
// xt tiles arrive via TDM (tensor_load_to_lds), double-buffered.
__global__ __launch_bounds__(512, 1) void k_lstm(const __bf16* __restrict__ seqbf,
                                                 float* __restrict__ outg,
                                                 const __bf16* __restrict__ wcat_g,
                                                 const float* __restrict__ bih,
                                                 const float* __restrict__ bhh) {
  extern __shared__ char smem[];
  __bf16* Wc  = (__bf16*)smem;                  // [0, 262144)
  __bf16* Xh  = (__bf16*)(smem + 262144);       // [262144, 278528)
  __bf16* Xx0 = (__bf16*)(smem + 278528);       // [278528, 294912)
  __bf16* Xx1 = (__bf16*)(smem + 294912);       // [294912, 311296)
  const int tid  = threadIdx.x;
  const int lane = tid & 31, wave = tid >> 5;
  const int half = lane >> 4, l15 = lane & 15;

  { // stage weights into LDS once (16384 uint4)
    const uint4* src = (const uint4*)wcat_g;
    uint4* dst = (uint4*)Wc;
    for (int i = tid; i < 16384; i += 512) dst[i] = src[i];
  }
  for (int i = tid; i < 64 * A_; i += 512) {
    Xh[i] = f2bf(0.f); Xx0[i] = f2bf(0.f); Xx1[i] = f2bf(0.f);
  }

  // per-lane persistent state: 2 tile groups, 8 cells each
  float creg[2][8];
  float bi_[2], bf_[2], bg_[2], bo_[2];
  int mt_[2], ic0_[2];
#pragma unroll
  for (int s = 0; s < 2; ++s) {
    const int g = wave * 2 + s;
    mt_[s]  = g >> 3;
    ic0_[s] = (g & 7) * 16;
    const int col = ic0_[s] + l15;
    bi_[s] = bih[col]       + bhh[col];
    bf_[s] = bih[col + 128] + bhh[col + 128];
    bg_[s] = bih[col + 256] + bhh[col + 256];
    bo_[s] = bih[col + 384] + bhh[col + 384];
#pragma unroll
    for (int v = 0; v < 8; ++v) creg[s][v] = 0.f;
  }
  __syncthreads();
  if (wave == 0) tdm_load_xt(seqbf, (uint32_t)(uintptr_t)Xx0);

  for (int n = 0; n < N_; ++n) {
    // prefetch next xt into the other buffer (its last readers finished at the
    // post-GEMM barrier of step n-1), then wait for xt(n); tensor ops complete
    // in order, so tensorcnt<=1 means load(n) is done.
    if (wave == 0) {
      if (n + 1 < N_) {
        tdm_load_xt(seqbf + (size_t)(n + 1) * TA_,
                    (uint32_t)(uintptr_t)(((n + 1) & 1) ? Xx1 : Xx0));
        __builtin_amdgcn_s_wait_tensorcnt(1);
      } else {
        __builtin_amdgcn_s_wait_tensorcnt(0);
      }
    }
    __syncthreads();  // xt(n) + prev-step h updates visible to all waves
    const __bf16* Xc = (n & 1) ? Xx1 : Xx0;

    v8f acc[2][4];
#pragma unroll
    for (int s = 0; s < 2; ++s)
#pragma unroll
      for (int q = 0; q < 4; ++q) acc[s][q] = zero8();

    for (int s = 0; s < 2; ++s) {
      const int mt = mt_[s], ic0 = ic0_[s];
      for (int kc = 0; kc < 4; ++kc) {      // x part: K 0..127
        const int k0 = kc * 32;
        v16bf Af = load_a_frag(Xc, A_, mt * 16, k0);
#pragma unroll
        for (int q = 0; q < 4; ++q)
          acc[s][q] = wmma_bf16(Af, load_b_frag_wt(Wc, KC_, ic0 + q * A_, k0), acc[s][q]);
      }
      for (int kc = 0; kc < 4; ++kc) {      // h part: K 128..255
        const int k0 = kc * 32;
        v16bf Af = load_a_frag(Xh, A_, mt * 16, k0);
#pragma unroll
        for (int q = 0; q < 4; ++q)
          acc[s][q] = wmma_bf16(Af, load_b_frag_wt(Wc, KC_, ic0 + q * A_, 128 + k0), acc[s][q]);
      }
    }
    __syncthreads();  // all GEMM reads of Xh done before cell update rewrites it

    float* sp = outg + (size_t)n * TA_;
    for (int s = 0; s < 2; ++s) {
      const int mt = mt_[s], col = ic0_[s] + l15;
#pragma unroll
      for (int v = 0; v < 8; ++v) {
        const int row = mt * 16 + v + half * 8;
        if (row < T_) {
          float gi = acc[s][0][v] + bi_[s];
          float gf = acc[s][1][v] + bf_[s];
          float gg = acc[s][2][v] + bg_[s];
          float go = acc[s][3][v] + bo_[s];
          float cn = sigm(gf) * creg[s][v] + sigm(gi) * tanhf(gg);
          float hh = sigm(go) * tanhf(cn);
          creg[s][v] = cn;
          Xh[row * A_ + col] = f2bf(hh);   // h-half for next step
          sp[row * A_ + col] = hh;         // out[n]
        }
      }
    }
    // next iteration's post-wait barrier orders these updates vs next GEMM
  }
}

// ---- fused per-sample tail: cov -> Q -> rets -> power iter -> FISTA -------
__global__ __launch_bounds__(256, 2) void k_fused(const float* __restrict__ outg,
                                                  const __bf16* __restrict__ wlb,
                                                  const float* __restrict__ csum,
                                                  const float* __restrict__ blin,
                                                  float* __restrict__ dout) {
  extern __shared__ char smem[];
  __bf16* mB   = (__bf16*)smem;                 // [0,16384)
  __bf16* covB = (__bf16*)(smem + 16384);       // [16384,49152)
  float*  Qs   = (float*)(smem + 49152);        // [49152,114688)
  float*  mean = (float*)(smem + 114688);       // 128
  float*  rets = (float*)(smem + 115200);       // 128
  float*  vec  = (float*)(smem + 115712);       // 128
  float*  zv   = (float*)(smem + 116224);       // 128
  float*  red  = (float*)(smem + 116736);       // 256
  const int n = blockIdx.x, tid = threadIdx.x;
  const int lane = tid & 31, wave = tid >> 5, half = lane >> 4, l15 = lane & 15;
  const float* on = outg + (size_t)n * TA_;

  if (tid < A_) {
    float s = 0.f;
    for (int t = 0; t < T_; ++t) s += on[t * A_ + tid];
    mean[tid] = s * (1.0f / T_);
  }
  __syncthreads();
  for (int idx = tid; idx < 64 * A_; idx += 256) {
    int t = idx >> 7, a = idx & 127;
    mB[idx] = f2bf((t < T_) ? (on[idx] - mean[a]) : 0.f);
  }
  __syncthreads();

  // cov = shrink(m^T m / (T-1)) via WMMA, K=64 (zero-padded)
  for (int jt = 0; jt < 8; ++jt) {
    v8f acc = zero8();
    for (int kc = 0; kc < 2; ++kc) {
      const int k0 = kc * 32;
      acc = wmma_bf16(load_a_frag_tr(mB, A_, wave * 16, k0),
                      load_b_frag_rm(mB, A_, jt * 16, k0), acc);
    }
    const int col = jt * 16 + l15;
#pragma unroll
    for (int v = 0; v < 8; ++v) {
      const int row = wave * 16 + v + half * 8;
      float s = acc[v] * (1.0f / (T_ - 1));
      covB[row * A_ + col] = f2bf((row == col) ? s : 0.5f * s);
    }
  }
  __syncthreads();

  // Q = cov * cov (symmetric) via WMMA, K=128
  for (int jt = 0; jt < 8; ++jt) {
    v8f acc = zero8();
    for (int kc = 0; kc < 4; ++kc) {
      const int k0 = kc * 32;
      acc = wmma_bf16(load_a_frag(covB, A_, wave * 16, k0),
                      load_b_frag_rm(covB, A_, jt * 16, k0), acc);
    }
    const int col = jt * 16 + l15;
#pragma unroll
    for (int v = 0; v < 8; ++v)
      Qs[(wave * 16 + v + half * 8) * A_ + col] = acc[v];
  }
  __syncthreads();

  // exp_rets = tanh(out_flat @ W_lin^T + b) via centered m + mean*colsum(W_lin)
  {
    const int j = tid >> 1, hh2 = tid & 1;
    const __bf16* wrow = wlb + (size_t)j * TA_;
    float s = 0.f;
    for (int t = hh2 * 25; t < hh2 * 25 + 25; ++t) {
      const __bf16* wp = wrow + t * A_;
      __builtin_prefetch(wp + A_, 0, 0);
      const __bf16* mp = mB + t * A_;
#pragma unroll 4
      for (int a = 0; a < A_; ++a) s += (float)mp[a] * (float)wp[a];
    }
    red[tid] = s;
    __syncthreads();
    if (hh2 == 0) {
      float tot = red[2 * j] + red[2 * j + 1];
      const float* cp = csum + j * A_;
      float corr = 0.f;
      for (int a = 0; a < A_; ++a) corr += mean[a] * cp[a];
      rets[j] = tanhf(tot + corr + blin[j]);
    }
  }
  __syncthreads();
  if (tid >= 32) return;   // wave 0 finishes the sample barrier-free

  int rows[4];
#pragma unroll
  for (int j = 0; j < 4; ++j) { rows[j] = lane + 32 * j; vec[rows[j]] = 1.0f; }

  const float4* vv = (const float4*)vec;
  const float4* zz = (const float4*)zv;

  for (int it = 0; it < 30; ++it) {
    float y[4];
#pragma unroll
    for (int j = 0; j < 4; ++j) {
      const float4* qr = (const float4*)(Qs + rows[j] * A_);
      float s = 0.f;
      for (int b = 0; b < 32; ++b) {
        float4 q = qr[b], x = vv[b];
        s += q.x * x.x + q.y * x.y + q.z * x.z + q.w * x.w;
      }
      y[j] = s;
    }
    float nr = wsum(y[0] * y[0] + y[1] * y[1] + y[2] * y[2] + y[3] * y[3]);
    float rinv = 1.0f / (sqrtf(nr) + 1e-12f);
#pragma unroll
    for (int j = 0; j < 4; ++j) vec[rows[j]] = y[j] * rinv;
  }
  float lam;
  {
    float p = 0.f;
#pragma unroll
    for (int j = 0; j < 4; ++j) {
      const float4* qr = (const float4*)(Qs + rows[j] * A_);
      float s = 0.f;
      for (int b = 0; b < 32; ++b) {
        float4 q = qr[b], x = vv[b];
        s += q.x * x.x + q.y * x.y + q.z * x.z + q.w * x.w;
      }
      p += vec[rows[j]] * s;
    }
    lam = wsum(p);
  }
  const float stp = 1.0f / (2.0f * lam + 1e-6f);

  float rr[4], w[4], z[4];
#pragma unroll
  for (int j = 0; j < 4; ++j) {
    rr[j] = rets[rows[j]];
    w[j] = z[j] = 1.0f / A_;
    zv[rows[j]] = z[j];
  }
  float tt = 1.0f;
  for (int it = 0; it < 200; ++it) {
    float u[4];
#pragma unroll
    for (int j = 0; j < 4; ++j) {
      const float4* qr = (const float4*)(Qs + rows[j] * A_);
      float s = 0.f;
      for (int b = 0; b < 32; ++b) {
        float4 q = qr[b], x = zz[b];
        s += q.x * x.x + q.y * x.y + q.z * x.z + q.w * x.w;
      }
      u[j] = z[j] - stp * (2.0f * s - rr[j]);
    }
    float lo = wminr(fminf(fminf(u[0], u[1]), fminf(u[2], u[3]))) - 1.0f;
    float hi = wmaxr(fmaxf(fmaxf(u[0], u[1]), fmaxf(u[2], u[3]))) + 1.0f;
    for (int bi = 0; bi < 60; ++bi) {
      float mid = 0.5f * (lo + hi);
      float rs = clip1(u[0] - mid) + clip1(u[1] - mid) + clip1(u[2] - mid) + clip1(u[3] - mid);
      rs = wsum(rs) - 1.0f;
      if (rs > 0.0f) lo = mid; else hi = mid;   // uniform across wave
    }
    float tau = 0.5f * (lo + hi);
    float act = 0.f, rsm = 0.f;
#pragma unroll
    for (int j = 0; j < 4; ++j) {
      float d = u[j] - tau;
      act += (d > -1.0f && d < 1.0f) ? 1.0f : 0.0f;
      rsm += clip1(d);
    }
    act = wsum(act);
    rsm = wsum(rsm) - 1.0f;
    tau += rsm / fmaxf(act, 1.0f);
    float tn = 0.5f * (1.0f + sqrtf(1.0f + 4.0f * tt * tt));
#pragma unroll
    for (int j = 0; j < 4; ++j) {
      float wn = clip1(u[j] - tau);
      z[j] = wn + ((tt - 1.0f) / tn) * (wn - w[j]);
      w[j] = wn;
      zv[rows[j]] = z[j];
    }
    tt = tn;
  }
#pragma unroll
  for (int j = 0; j < 4; ++j) dout[n * A_ + rows[j]] = w[j];
}

// ---- host ------------------------------------------------------------------
extern "C" void kernel_launch(void* const* d_in, const int* in_sizes, int n_in,
                              void* d_out, int out_size, void* d_ws, size_t ws_size,
                              hipStream_t stream) {
  (void)in_sizes; (void)n_in; (void)out_size; (void)ws_size;
  const float* x     = (const float*)d_in[0];
  const float* gamma = (const float*)d_in[1];
  const float* beta  = (const float*)d_in[2];
  const float* wih   = (const float*)d_in[3];
  const float* whh   = (const float*)d_in[4];
  const float* bih   = (const float*)d_in[5];
  const float* bhh   = (const float*)d_in[6];
  const float* wlin  = (const float*)d_in[7];
  const float* blin  = (const float*)d_in[8];
  float* dout = (float*)d_out;

  char* ws = (char*)d_ws;
  float*  outg  = (float*)ws;                                      // 26,214,400 B
  __bf16* seqbf = (__bf16*)(ws + 26214400);                        // 13,107,200 B
  __bf16* wcat  = (__bf16*)(ws + 26214400 + 13107200);             //    262,144 B
  __bf16* wlb   = (__bf16*)(ws + 26214400 + 13107200 + 262144);    //  1,638,400 B
  float*  csum  = (float*)(ws + 26214400 + 13107200 + 262144 + 1638400); // 65,536 B

  hipFuncSetAttribute((const void*)k_lstm,  hipFuncAttributeMaxDynamicSharedMemorySize, 311296);
  hipFuncSetAttribute((const void*)k_fused, hipFuncAttributeMaxDynamicSharedMemorySize, 117760);

  k_prep_wcat<<<512, 256, 0, stream>>>(wih, whh, wcat);
  k_prep_wlin<<<3200, 256, 0, stream>>>(wlin, wlb);
  k_prep_csum<<<64, 256, 0, stream>>>(wlin, csum);
  k_layernorm<<<N_, 256, 0, stream>>>(x, gamma, beta, seqbf);
  k_lstm<<<1, 512, 311296, stream>>>(seqbf, outg, wcat, bih, bhh);
  k_fused<<<N_, 256, 117760, stream>>>(outg, wlb, csum, blin, dout);
}
